// EdgeNet_22771916604198
// MI455X (gfx1250) — matrix operand order repaired and checked by reading, not compiled, not run on hardware
//
#include <hip/hip_runtime.h>
#include <stdint.h>

#define NN       512
#define NPAIRS   262144      // 512*512
#define F_DIM    256
#define C1_DIM   256
#define C2_DIM   128
#define LEAKY    0.01f
#define BN_EPS   1e-5f

typedef __attribute__((ext_vector_type(16))) __bf16 v16bf;
typedef __attribute__((ext_vector_type(8)))  float  v8f;

struct Frag32B { uint4 lo, hi; };   // 32 bytes == 16 bf16 == one WMMA A/B fragment

// ---- bf16 helpers: use native HW converts (v_cvt_*bf16*) ----
static __device__ __forceinline__ unsigned short f2bf(float x) {
  __bf16 h = (__bf16)x;
  return __builtin_bit_cast(unsigned short, h);
}
static __device__ __forceinline__ float bf2f(unsigned short u) {
  __bf16 h = __builtin_bit_cast(__bf16, u);
  return (float)h;
}
static __device__ __forceinline__ unsigned int pk2(float a, float b) {
  return (unsigned int)f2bf(a) | ((unsigned int)f2bf(b) << 16);
}

// ---- WMMA fragment loads from LDS (row-major tiles, ldh halves/row) ----
// A (16x32 bf16): lane<16 -> row=lane,    K = {0..7} then {16..23}
//                 lane>=16 -> row=lane-16, K = {8..15} then {24..31}
static __device__ __forceinline__ v16bf load_a_frag(const unsigned short* sm, int ldh) {
  const int lane = threadIdx.x & 31;
  const int hs = lane >> 4, r = lane & 15;
  const unsigned short* p = sm + r * ldh + hs * 8;
  Frag32B f;
  f.lo = *(const uint4*)(p);        // 8 halves
  f.hi = *(const uint4*)(p + 16);   // next K-chunk (+16 halves)
  return __builtin_bit_cast(v16bf, f);
}
// B (32x16 bf16) staged as [n][k] (K contiguous per column):
//   lane<16 -> col=lane, K={0..7} then {16..23} ; lane>=16 -> col=lane-16, K={8..15} then {24..31}
static __device__ __forceinline__ v16bf load_b_frag(const unsigned short* sm, int ldh) {
  const int lane = threadIdx.x & 31;
  const int hs = lane >> 4, n = lane & 15;
  const unsigned short* p = sm + n * ldh + hs * 8;
  Frag32B f;
  f.lo = *(const uint4*)(p);
  f.hi = *(const uint4*)(p + 16);
  return __builtin_bit_cast(v16bf, f);
}

static __device__ __forceinline__ v8f wmma_bf16(v16bf a, v16bf b, v8f c) {
  return __builtin_amdgcn_wmma_f32_16x16x32_bf16(false, a, false, b, (short)0, c, false, false);
}

// =====================================================================
// GEMM1: h0[c,p] = sum_f W0[c,f] * |feat[i_p,f] - feat[j_p,f]|
// grid = (1024, 4), block = 256 (8 waves = 2M x 4N).
// WG tile: 64 channels x 256 pairs, K=256. Wave tile: 32ch x 64pairs.
// =====================================================================
__global__ __launch_bounds__(256) void gemm1_kernel(
    const float* __restrict__ feat, const float* __restrict__ W0,
    unsigned short* __restrict__ h0)
{
  __shared__ __align__(16) unsigned short As[64 * 32];    //  4 KB
  __shared__ __align__(16) unsigned short Bs[256 * 32];   // 16 KB

  const int tid   = threadIdx.x;
  const int mblk  = blockIdx.y * 64;
  const int pbase = blockIdx.x * 256;
  const int irow  = pbase >> 9;      // constant within WG (256 | 512)
  const int j0    = pbase & 511;

  const int w  = tid >> 5;
  const int wm = w & 1;              // 2 waves along M (32 ch each)
  const int wn = w >> 1;             // 4 waves along N (64 pairs each)

  v8f zero = {0.f,0.f,0.f,0.f,0.f,0.f,0.f,0.f};
  v8f acc[2][4];
  for (int mt = 0; mt < 2; ++mt)
    for (int nt = 0; nt < 4; ++nt) acc[mt][nt] = zero;

  for (int kb = 0; kb < F_DIM; kb += 32) {
    // ---- stage A: W0 tile 64x32 -> bf16 LDS (threads 0..127) ----
    if (tid < 128) {
      const int ch = tid >> 1;
      const int kc = (tid & 1) * 16;
      const float* src = W0 + (mblk + ch) * F_DIM + kb + kc;
      if (kb + 32 < F_DIM) __builtin_prefetch(src + 32, 0, 0);
      float4 f0 = *(const float4*)(src);
      float4 f1 = *(const float4*)(src + 4);
      float4 f2 = *(const float4*)(src + 8);
      float4 f3 = *(const float4*)(src + 12);
      uint4 o0, o1;
      o0.x = pk2(f0.x, f0.y); o0.y = pk2(f0.z, f0.w);
      o0.z = pk2(f1.x, f1.y); o0.w = pk2(f1.z, f1.w);
      o1.x = pk2(f2.x, f2.y); o1.y = pk2(f2.z, f2.w);
      o1.z = pk2(f3.x, f3.y); o1.w = pk2(f3.z, f3.w);
      unsigned short* dst = As + ch * 32 + kc;
      *(uint4*)(dst)     = o0;
      *(uint4*)(dst + 8) = o1;
    }
    // ---- stage B: |feat_i - feat_j| tile 256x32 -> bf16 LDS (one pair/thread) ----
    {
      const int pp = tid;
      const float* fi = feat + irow * F_DIM + kb;
      const float* fj = feat + (j0 + pp) * F_DIM + kb;
      unsigned short* dst = Bs + pp * 32;
      for (int c4 = 0; c4 < 4; ++c4) {           // 4 chunks of 8 floats
        float4 a0 = *(const float4*)(fi + c4 * 8);
        float4 a1 = *(const float4*)(fi + c4 * 8 + 4);
        float4 b0 = *(const float4*)(fj + c4 * 8);
        float4 b1 = *(const float4*)(fj + c4 * 8 + 4);
        uint4 o;
        o.x = pk2(fabsf(a0.x - b0.x), fabsf(a0.y - b0.y));
        o.y = pk2(fabsf(a0.z - b0.z), fabsf(a0.w - b0.w));
        o.z = pk2(fabsf(a1.x - b1.x), fabsf(a1.y - b1.y));
        o.w = pk2(fabsf(a1.z - b1.z), fabsf(a1.w - b1.w));
        *(uint4*)(dst + c4 * 8) = o;
      }
    }
    __syncthreads();

    v16bf af0 = load_a_frag(As + (wm * 32 +  0) * 32, 32);
    v16bf af1 = load_a_frag(As + (wm * 32 + 16) * 32, 32);
    for (int nt = 0; nt < 4; ++nt) {
      v16bf bfr = load_b_frag(Bs + (wn * 64 + nt * 16) * 32, 32);
      acc[0][nt] = wmma_bf16(af0, bfr, acc[0][nt]);
      acc[1][nt] = wmma_bf16(af1, bfr, acc[1][nt]);
    }
    __syncthreads();
  }

  // Epilogue: store pre-BN h0 as bf16. D layout: vgpr e -> M = hs*8+e, N = lane&15
  const int lane = tid & 31;
  const int hs = lane >> 4, nl = lane & 15;
  for (int mt = 0; mt < 2; ++mt)
    for (int nt = 0; nt < 4; ++nt) {
      const int cbase = mblk + wm * 32 + mt * 16 + hs * 8;
      const int p = pbase + wn * 64 + nt * 16 + nl;
      for (int e = 0; e < 8; ++e)
        h0[(size_t)(cbase + e) * NPAIRS + p] = f2bf(acc[mt][nt][e]);
    }
}

// =====================================================================
// BN stats: one block per channel; produce affine scale/shift
// =====================================================================
__global__ __launch_bounds__(256) void bn_stats_kernel(
    const unsigned short* __restrict__ h, const float* __restrict__ gamma,
    const float* __restrict__ beta, float* __restrict__ s, float* __restrict__ t)
{
  const int c = blockIdx.x;
  const unsigned short* row = h + (size_t)c * NPAIRS;
  float sum = 0.f, sq = 0.f;
  for (int base = threadIdx.x * 8; base < NPAIRS; base += 256 * 8) {
    uint4 u = *(const uint4*)(row + base);
    unsigned int wds[4] = {u.x, u.y, u.z, u.w};
    for (int q = 0; q < 4; ++q) {
      float lo = bf2f((unsigned short)(wds[q] & 0xffffu));
      float hi = bf2f((unsigned short)(wds[q] >> 16));
      sum += lo + hi;
      sq  += lo * lo + hi * hi;
    }
  }
  __shared__ float rs[256], rq[256];
  rs[threadIdx.x] = sum; rq[threadIdx.x] = sq;
  __syncthreads();
  for (int off = 128; off > 0; off >>= 1) {
    if ((int)threadIdx.x < off) {
      rs[threadIdx.x] += rs[threadIdx.x + off];
      rq[threadIdx.x] += rq[threadIdx.x + off];
    }
    __syncthreads();
  }
  if (threadIdx.x == 0) {
    float mu  = rs[0] * (1.f / (float)NPAIRS);
    float var = rq[0] * (1.f / (float)NPAIRS) - mu * mu;
    float sc  = gamma[c] * rsqrtf(var + BN_EPS);
    s[c] = sc;
    t[c] = beta[c] - mu * sc;
  }
}

// =====================================================================
// GEMM2: h1[c2,p] = sum_c1 W1[c2,c1] * leaky(s0[c1]*h0[c1,p]+t0[c1])
// grid = (1024, 2), block = 256 (8 waves = 2M x 4N)
// =====================================================================
__global__ __launch_bounds__(256) void gemm2_kernel(
    const unsigned short* __restrict__ h0, const float* __restrict__ W1,
    const float* __restrict__ s0, const float* __restrict__ t0,
    unsigned short* __restrict__ h1)
{
  __shared__ __align__(16) unsigned short As[64 * 32];    //  4 KB
  __shared__ __align__(16) unsigned short Bs[256 * 32];   // 16 KB

  const int tid   = threadIdx.x;
  const int mblk  = blockIdx.y * 64;
  const int pbase = blockIdx.x * 256;

  const int w  = tid >> 5;
  const int wm = w & 1;
  const int wn = w >> 1;

  v8f zero = {0.f,0.f,0.f,0.f,0.f,0.f,0.f,0.f};
  v8f acc[2][4];
  for (int mt = 0; mt < 2; ++mt)
    for (int nt = 0; nt < 4; ++nt) acc[mt][nt] = zero;

  for (int kb = 0; kb < C1_DIM; kb += 32) {
    // ---- stage A: W1 tile 64x32 ----
    if (tid < 128) {
      const int ch = tid >> 1;
      const int kc = (tid & 1) * 16;
      const float* src = W1 + (mblk + ch) * C1_DIM + kb + kc;
      float4 f0 = *(const float4*)(src);
      float4 f1 = *(const float4*)(src + 4);
      float4 f2 = *(const float4*)(src + 8);
      float4 f3 = *(const float4*)(src + 12);
      uint4 o0, o1;
      o0.x = pk2(f0.x, f0.y); o0.y = pk2(f0.z, f0.w);
      o0.z = pk2(f1.x, f1.y); o0.w = pk2(f1.z, f1.w);
      o1.x = pk2(f2.x, f2.y); o1.y = pk2(f2.z, f2.w);
      o1.z = pk2(f3.x, f3.y); o1.w = pk2(f3.z, f3.w);
      unsigned short* dst = As + ch * 32 + kc;
      *(uint4*)(dst)     = o0;
      *(uint4*)(dst + 8) = o1;
    }
    // ---- stage B: act(bn(h0)) tile 256x32 (one pair/thread) ----
    {
      const int pp = tid;
      unsigned short* dst = Bs + pp * 32;
      for (int c8 = 0; c8 < 4; ++c8) {
        float v[8];
        for (int q = 0; q < 8; ++q) {
          const int c = kb + c8 * 8 + q;
          float x = bf2f(h0[(size_t)c * NPAIRS + pbase + pp]) * s0[c] + t0[c];
          v[q] = (x > 0.f) ? x : LEAKY * x;
        }
        uint4 o;
        o.x = pk2(v[0], v[1]); o.y = pk2(v[2], v[3]);
        o.z = pk2(v[4], v[5]); o.w = pk2(v[6], v[7]);
        *(uint4*)(dst + c8 * 8) = o;
      }
    }
    __syncthreads();

    v16bf af0 = load_a_frag(As + (wm * 32 +  0) * 32, 32);
    v16bf af1 = load_a_frag(As + (wm * 32 + 16) * 32, 32);
    for (int nt = 0; nt < 4; ++nt) {
      v16bf bfr = load_b_frag(Bs + (wn * 64 + nt * 16) * 32, 32);
      acc[0][nt] = wmma_bf16(af0, bfr, acc[0][nt]);
      acc[1][nt] = wmma_bf16(af1, bfr, acc[1][nt]);
    }
    __syncthreads();
  }

  const int lane = tid & 31;
  const int hs = lane >> 4, nl = lane & 15;
  for (int mt = 0; mt < 2; ++mt)
    for (int nt = 0; nt < 4; ++nt) {
      const int cbase = mblk + wm * 32 + mt * 16 + hs * 8;
      const int p = pbase + wn * 64 + nt * 16 + nl;
      for (int e = 0; e < 8; ++e)
        h1[(size_t)(cbase + e) * NPAIRS + p] = f2bf(acc[mt][nt][e]);
    }
}

// =====================================================================
// Head: logits -> sigmoid -> sim_val
// =====================================================================
__global__ __launch_bounds__(256) void head_kernel(
    const unsigned short* __restrict__ h1, const float* __restrict__ s1,
    const float* __restrict__ t1, const float* __restrict__ Wout,
    const float* __restrict__ bout, float* __restrict__ sim)
{
  const int p = blockIdx.x * 256 + threadIdx.x;
  float a = bout[0];
  for (int c = 0; c < C2_DIM; ++c) {
    float v = bf2f(h1[(size_t)c * NPAIRS + p]) * s1[c] + t1[c];
    v = (v > 0.f) ? v : LEAKY * v;
    a += Wout[c] * v;
  }
  sim[p] = 1.f / (1.f + __expf(-a));
}

// =====================================================================
// Column sums of numerator  nm(i,j) = same_label*sim + eye + 1e-6
// (sum over i, per j). grid = 4, block = 128; reads are row-coalesced.
// =====================================================================
__global__ __launch_bounds__(128) void colsum_kernel(
    const float* __restrict__ sim, const long long* __restrict__ lab,
    float* __restrict__ colsum)
{
  const int j = blockIdx.x * 128 + threadIdx.x;
  const long long lj = lab[j];
  float sum = 0.f;
  for (int i = 0; i < NN; ++i) {
    float nm = ((lab[i] == lj) ? sim[i * NN + j] : 0.f)
             + ((i == j) ? 1.f : 0.f) + 1e-6f;
    sum += nm;
  }
  colsum[j] = sum;
}

// =====================================================================
// edge_feat(i,j) = nm(i,j) / colsum[j]
// =====================================================================
__global__ __launch_bounds__(256) void edge_kernel(
    const float* __restrict__ sim, const long long* __restrict__ lab,
    const float* __restrict__ colsum, float* __restrict__ out)
{
  const int p = blockIdx.x * 256 + threadIdx.x;
  const int i = p >> 9, j = p & 511;
  float nm = ((lab[i] == lab[j]) ? sim[p] : 0.f)
           + ((i == j) ? 1.f : 0.f) + 1e-6f;
  out[p] = nm / colsum[j];
}

// =====================================================================
extern "C" void kernel_launch(void* const* d_in, const int* in_sizes, int n_in,
                              void* d_out, int out_size, void* d_ws, size_t ws_size,
                              hipStream_t stream)
{
  (void)in_sizes; (void)n_in; (void)out_size; (void)ws_size;
  const float*      feat = (const float*)d_in[0];       // 512x256
  const long long*  lab  = (const long long*)d_in[1];   // 512 (int64)
  const float*      W0   = (const float*)d_in[2];       // 256x256
  const float*      g0   = (const float*)d_in[3];
  const float*      b0   = (const float*)d_in[4];
  const float*      W1   = (const float*)d_in[5];       // 128x256
  const float*      g1   = (const float*)d_in[6];
  const float*      b1   = (const float*)d_in[7];
  const float*      Wout = (const float*)d_in[8];       // 128
  const float*      bout = (const float*)d_in[9];       // 1

  float* edge_out = (float*)d_out;             // 262144 (edge_feat)
  float* sim_out  = (float*)d_out + NPAIRS;    // 262144 (sim_val)

  char* ws = (char*)d_ws;
  unsigned short* h0 = (unsigned short*)ws;                                    // 128 MB
  unsigned short* h1 = (unsigned short*)(ws + (size_t)C1_DIM * NPAIRS * 2);    //  64 MB
  float* sp = (float*)(ws + (size_t)(C1_DIM + C2_DIM) * NPAIRS * 2);
  float* s0 = sp, *t0 = sp + 256, *s1 = sp + 512, *t1 = sp + 640, *colsum = sp + 768;

  gemm1_kernel   <<<dim3(NPAIRS / 256, C1_DIM / 64), 256, 0, stream>>>(feat, W0, h0);
  bn_stats_kernel<<<C1_DIM, 256, 0, stream>>>(h0, g0, b0, s0, t0);
  gemm2_kernel   <<<dim3(NPAIRS / 256, C2_DIM / 64), 256, 0, stream>>>(h0, W1, s0, t0, h1);
  bn_stats_kernel<<<C2_DIM, 256, 0, stream>>>(h1, g1, b1, s1, t1);
  head_kernel    <<<NPAIRS / 256, 256, 0, stream>>>(h1, s1, t1, Wout, bout, sim_out);
  colsum_kernel  <<<NN / 128, 128, 0, stream>>>(sim_out, lab, colsum);
  edge_kernel    <<<NPAIRS / 256, 256, 0, stream>>>(sim_out, lab, colsum, edge_out);
}
